// Gemma4Router_30288109371938
// MI455X (gfx1250) — matrix operand order repaired and checked by reading, not compiled, not run on hardware
//
#include <hip/hip_runtime.h>
#include <hip/hip_bf16.h>
#include <math.h>
#include <stdint.h>

#define HIDDEN 2816
#define NEXP   128
#define TOPK   8
#define TOKENS 32768
#define TPB    8                  // tokens per block
#define KTILE  32
#define NKT    (HIDDEN / KTILE)   // 88 K-steps
#define SA_STRIDE (HIDDEN + 8)    // bf16 elems; +8 pads LDS rows off bank alignment
#define SROW   (NEXP + 4)

typedef __attribute__((ext_vector_type(16))) __bf16 v16bf;
typedef __attribute__((ext_vector_type(4)))  __bf16 bf16x4;
typedef __attribute__((ext_vector_type(8)))  float  v8f;
typedef __attribute__((ext_vector_type(4)))  unsigned int v4u;
typedef __attribute__((ext_vector_type(8)))  int    v8i;
typedef __attribute__((ext_vector_type(4)))  int    v4i;

union FragBF { uint4 u[2]; v16bf v; };

// Tensor Data Mover: toolchain arity hedge (5-arg ROCm 7.2, 6-arg amdgpu-toolchain).
#if __has_include(<hip/amd_detail/amd_gfx1250_TDM.h>)
#define TDM_LOAD(g0, g1, g2, g3) \
    __builtin_amdgcn_tensor_load_to_lds((g0), (g1), (g2), (g3), (v8i){0,0,0,0,0,0,0,0}, 0)
#else
#define TDM_LOAD(g0, g1, g2, g3) \
    __builtin_amdgcn_tensor_load_to_lds((g0), (g1), (g2), (g3), 0)
#endif

// One-time prep: fold scale * H^-0.5 into W and convert to bf16 (stays L2-resident).
__global__ void pack_w_kernel(const float* __restrict__ W,
                              const float* __restrict__ scale,
                              __bf16* __restrict__ Wb) {
    int i = blockIdx.x * blockDim.x + threadIdx.x;
    if (i < NEXP * HIDDEN) {
        int h = i % HIDDEN;
        float rs = rsqrtf((float)HIDDEN);
        Wb[i] = (__bf16)(W[i] * scale[h] * rs);
    }
}

__global__ __launch_bounds__(256)
void router_kernel(const float* __restrict__ x,
                   const __bf16* __restrict__ Wb,
                   float* __restrict__ outw,
                   int* __restrict__ outi) {
    __shared__ float  sX[TPB * HIDDEN];      // TDM-staged raw fp32 activations
    __shared__ __bf16 sA[TPB * SA_STRIDE];   // normalized bf16 activations
    __shared__ float  sS[TPB * SROW];        // 8 x 128 logits

    const int tid  = threadIdx.x;
    const int lane = tid & 31;
    const int wv   = tid >> 5;               // wave id 0..7 (wave32); token id == wv

    // ---------- Phase 0: per-wave TDM DMA of this wave's token row into LDS ----------
    {
        const unsigned long long gaddr =
            (unsigned long long)(const void*)(x + ((size_t)blockIdx.x * TPB + wv) * HIDDEN);
        const unsigned ldsoff =
            (unsigned)(uintptr_t)(void*)&sX[wv * HIDDEN];
        // D# group0: count=1 | lds_addr | global_addr[56:0] | type=2
        v4u g0 = { 1u, ldsoff, (unsigned)gaddr,
                   (unsigned)((gaddr >> 32) & 0x1FFFFFFu) | (2u << 30) };
        // D# group1: data_size=4B; tensor_dim0=tile_dim0=2816; tensor_dim1=tile_dim1=1;
        //            tensor_dim0_stride=2816 (fields per ISA 8.4 bit layout)
        v8i g1 = { 0x00020000,                 // workgroup_mask=0, data_size=2 (4B)
                   (int)0x0B000000,            // tensor_dim0 lo16 (2816) << 16
                   (int)0x00010000,            // tensor_dim0 hi16 | tensor_dim1 lo16 (1)
                   (int)0x0B000000,            // tensor_dim1 hi16 | tile_dim0 (2816)
                   1,                          // tile_dim1=1, tile_dim2=0
                   2816,                       // tensor_dim0_stride lo32
                   0, 0 };
        v4i gz = { 0, 0, 0, 0 };
        TDM_LOAD(g0, g1, gz, gz);
        __builtin_amdgcn_s_wait_tensorcnt(0); // wait only this wave's DMA
    }

    // ---------- Phase 1: RMSNorm from LDS -> bf16 A row ----------
    {
        const float4* xr = (const float4*)&sX[wv * HIDDEN];
        float ss = 0.f;
        #pragma unroll 4
        for (int i = 0; i < HIDDEN / 4 / 32; ++i) {          // 22 iters
            float4 v = xr[i * 32 + lane];
            ss += v.x * v.x + v.y * v.y + v.z * v.z + v.w * v.w;
        }
        #pragma unroll
        for (int off = 16; off; off >>= 1) ss += __shfl_xor(ss, off, 32);
        const float r = rsqrtf(ss * (1.0f / HIDDEN) + 1e-6f);
        #pragma unroll 4
        for (int i = 0; i < HIDDEN / 4 / 32; ++i) {
            float4 v = xr[i * 32 + lane];
            bf16x4 o;
            o.x = (__bf16)(v.x * r); o.y = (__bf16)(v.y * r);
            o.z = (__bf16)(v.z * r); o.w = (__bf16)(v.w * r);
            *(bf16x4*)&sA[wv * SA_STRIDE + (i * 32 + lane) * 4] = o;
        }
    }
    __syncthreads();

    // ---------- Phase 2: 8x16 logits per wave via WMMA over 88 K-steps ----------
    // A rows M=8..15 duplicate M&7 (C row m depends only on A row m; dups ignored).
    v8f acc = {};
    const int lm = lane & 15;
    const int hi = (lane >> 4) & 1;
    const int arow = lm & 7;
    const __bf16* wrow = Wb + (size_t)(wv * 16 + lm) * HIDDEN;  // expert row (B: N=lm)
    for (int q = 0; q < NKT; ++q) {
        FragBF a, b;
        const int kA = q * KTILE + hi * 8;     // A: lane<16 K{0..7,16..23}, else {8..15,24..31}
        a.u[0] = *(const uint4*)&sA[arow * SA_STRIDE + kA];
        a.u[1] = *(const uint4*)&sA[arow * SA_STRIDE + kA + 16];
        const int kB = q * KTILE + hi * 8;     // B: lane<16 K{0..7,16..23}, else {8..15,24..31}
        b.u[0] = *(const uint4*)(wrow + kB);
        b.u[1] = *(const uint4*)(wrow + kB + 16);
        if (q + 1 < NKT) __builtin_prefetch(wrow + kB + KTILE, 0, 0);
        acc = __builtin_amdgcn_wmma_f32_16x16x32_bf16(false, a.v, false, b.v,
                                                      (short)0, acc, false, false);
    }
    if (lane < 16) {   // C: lanes 0-15 hold M=0..7 (VGPR r -> M=r), N=lane
        #pragma unroll
        for (int r = 0; r < 8; ++r)
            sS[r * SROW + wv * 16 + lane] = acc[r];
    }
    __syncthreads();

    // ---------- Phase 3: top-8 + softmax over winners (wave per token) ----------
    {
        const float NEG = -__builtin_inff();
        const float* row = &sS[wv * SROW];
        float s0 = row[lane], s1 = row[lane + 32], s2 = row[lane + 64], s3 = row[lane + 96];
        float v0 = 0.f, sum = 0.f, myv = 0.f;
        int myi = 0;
        #pragma unroll
        for (int t = 0; t < TOPK; ++t) {
            float v = s0; int i = lane;
            if (s1 > v) { v = s1; i = lane + 32; }
            if (s2 > v) { v = s2; i = lane + 64; }
            if (s3 > v) { v = s3; i = lane + 96; }
            #pragma unroll
            for (int off = 16; off; off >>= 1) {
                float ov = __shfl_xor(v, off, 32);
                int   oi = __shfl_xor(i, off, 32);
                if (ov > v || (ov == v && oi < i)) { v = ov; i = oi; }
            }
            if (t == 0) v0 = v;
            sum += __expf(v - v0);
            if (lane == t) { myv = v; myi = i; }
            if ((i & 31) == lane) {          // owner retires the winner
                const int j = i >> 5;
                if (j == 0) s0 = NEG; else if (j == 1) s1 = NEG;
                else if (j == 2) s2 = NEG; else s3 = NEG;
            }
        }
        if (lane < TOPK) {
            const size_t gt = (size_t)blockIdx.x * TPB + wv;
            outw[gt * TOPK + lane] = __expf(myv - v0) / sum;
            outi[gt * TOPK + lane] = myi;
        }
    }
}

extern "C" void kernel_launch(void* const* d_in, const int* in_sizes, int n_in,
                              void* d_out, int out_size, void* d_ws, size_t ws_size,
                              hipStream_t stream) {
    (void)in_sizes; (void)n_in; (void)out_size; (void)ws_size;
    const float* x     = (const float*)d_in[0];
    const float* W     = (const float*)d_in[1];
    const float* scale = (const float*)d_in[2];
    __bf16* Wb = (__bf16*)d_ws;                          // 128*2816 bf16 = 704 KB scratch
    float*  outw = (float*)d_out;                        // [32768*8] weights
    int*    outi = (int*)(outw + (size_t)TOKENS * TOPK); // [32768*8] indices

    pack_w_kernel<<<(NEXP * HIDDEN + 255) / 256, 256, 0, stream>>>(W, scale, Wb);
    router_kernel<<<TOKENS / TPB, 256, 0, stream>>>(x, Wb, outw, outi);
}
